// MultiHeadAttention_29394756174292
// MI455X (gfx1250) — matrix operand order repaired and checked by reading, not compiled, not run on hardware
//
#include <hip/hip_runtime.h>
#include <hip/hip_bf16.h>
#include <math.h>
#include <stdint.h>

typedef __bf16 bf16;
typedef __attribute__((ext_vector_type(16))) __bf16 v16bf;
typedef __attribute__((ext_vector_type(8)))  __bf16 v8bf;
typedef __attribute__((ext_vector_type(8)))  float  v8f;

#define DEV __device__ __forceinline__

constexpr int B_ = 4, S_ = 1024, DM_ = 1024, H_ = 16, DK_ = 64, DV_ = 64;
constexpr int E_ = H_ * DV_; // 1024 = concat-head width

DEV bf16 f2bf(float x) {
  unsigned u = __builtin_bit_cast(unsigned, x);
  unsigned r = u + 0x7FFFu + ((u >> 16) & 1u);   // round-to-nearest-even
  unsigned short h = (unsigned short)(r >> 16);
  return __builtin_bit_cast(bf16, h);
}

// Two contiguous 16-byte loads -> one 16-element bf16 fragment.
DEV v16bf frag16(const bf16* p0, const bf16* p1) {
  v8bf lo = *(const v8bf*)p0;
  v8bf hi = *(const v8bf*)p1;
  return __builtin_shufflevector(lo, hi, 0,1,2,3,4,5,6,7,8,9,10,11,12,13,14,15);
}
// A fragment (16x32, lane&15 = row); rp -> row's K-window start.
DEV v16bf fragA(const bf16* rp, int half) {
  return frag16(rp + half * 8, rp + 16 + half * 8);
}
// B fragment (32x16, lane&15 = col); rp -> B^T row's K-window start.
DEV v16bf fragB(const bf16* rp, int half) {
  return frag16(rp + half * 8, rp + 16 + half * 8);
}

DEV v8f wmma_bf16(v16bf a, v16bf b, v8f c) {
  return __builtin_amdgcn_wmma_f32_16x16x32_bf16(
      false, a, false, b, (short)0, c, false, false);
}

// ---- gfx1250 async LDS staging (tracked by ASYNCcnt) ----
DEV void async_ld16(unsigned lds_off, const bf16* g) {
  asm volatile("global_load_async_to_lds_b128 %0, %1, off"
               :: "v"(lds_off), "v"(g) : "memory");
}
DEV void wait_async_all() { asm volatile("s_wait_asynccnt 0x0" ::: "memory"); }
DEV void wait_async_16()  { asm volatile("s_wait_asynccnt 0x10" ::: "memory"); }
DEV unsigned lds_off_of(const void* p) { return (unsigned)(uintptr_t)p; }

// ---------------------------------------------------------------- conversions
__global__ void k_cvt(const float* __restrict__ src, bf16* __restrict__ dst, int n) {
  int i = blockIdx.x * blockDim.x + threadIdx.x;
  if (i < n) dst[i] = f2bf(src[i]);
}

// [H, DM, DK] f32 -> [H, DK, DM] bf16
__global__ void k_transpose_w(const float* __restrict__ src, bf16* __restrict__ dst) {
  int i = blockIdx.x * blockDim.x + threadIdx.x;
  if (i >= H_ * DM_ * DK_) return;
  int kk = i % DK_;
  int t  = i / DK_;
  int d  = t % DM_;
  int h  = t / DM_;
  dst[((size_t)h * DK_ + kk) * DM_ + d] = f2bf(src[i]);
}

// -------- 32x64 GEMM tile helpers: 2 A-frags, 4 B-frags, 8 accumulators ----
struct Frags { v16bf a[2]; v16bf b[4]; };

DEV void load_frags(Frags& f, const bf16* const* xr, const bf16* const* wr,
                    int d, int half) {
#pragma unroll
  for (int i = 0; i < 2; i++) f.a[i] = fragA(xr[i] + d, half);
#pragma unroll
  for (int j = 0; j < 4; j++) f.b[j] = fragB(wr[j] + d, half);
}
DEV void mma_frags(v8f acc[2][4], const Frags& f) {
#pragma unroll
  for (int i = 0; i < 2; i++)
#pragma unroll
    for (int j = 0; j < 4; j++)
      acc[i][j] = wmma_bf16(f.a[i], f.b[j], acc[i][j]);
}

// ---------------------------------------------------------------- QKV projection
// One wave computes a 32(s) x 64 output tile for one of Q/K/V of one (b,h).
// Peeled ping-pong pipeline: no in-loop branches, no fragment rotation moves.
__global__ void __launch_bounds__(32, 1)
k_qkv_proj(const bf16* __restrict__ xq, const bf16* __restrict__ xk,
           const bf16* __restrict__ xv,
           const bf16* __restrict__ Wqt, const bf16* __restrict__ Wkt,
           const bf16* __restrict__ Wvt,
           bf16* __restrict__ Qp, bf16* __restrict__ Kp, bf16* __restrict__ Vt) {
  int bid = blockIdx.x;
  int stile = bid & 31; bid >>= 5;  // S/32 = 32
  int h = bid & 15;     bid >>= 4;
  int b = bid & 3;
  int mat = bid >> 2;               // 0=Q 1=K 2=V

  const bf16* x = (mat == 0) ? xq : (mat == 1) ? xk : xv;
  const bf16* W = (mat == 0) ? Wqt : (mat == 1) ? Wkt : Wvt;

  int lane = threadIdx.x, half = lane >> 4, ncol = lane & 15;
  int s0 = stile * 32;

  const bf16* xr[2];
  const bf16* wr[4];
#pragma unroll
  for (int i = 0; i < 2; i++)
    xr[i] = x + ((size_t)b * S_ + s0 + i * 16 + ncol) * DM_;
#pragma unroll
  for (int j = 0; j < 4; j++)
    wr[j] = W + (size_t)h * DK_ * DM_ + (size_t)(j * 16 + ncol) * DM_;

  __builtin_prefetch(xr[0] + 512, 0, 0);   // global_prefetch_b8
  __builtin_prefetch(wr[0] + 512, 0, 0);

  v8f acc[2][4];
#pragma unroll
  for (int i = 0; i < 2; i++)
#pragma unroll
    for (int j = 0; j < 4; j++) acc[i][j] = {};

  Frags f0, f1;
  load_frags(f0, xr, wr, 0, half);
  for (int d = 0; d < DM_ - 64; d += 64) {
    load_frags(f1, xr, wr, d + 32, half);
    mma_frags(acc, f0);
    load_frags(f0, xr, wr, d + 64, half);
    mma_frags(acc, f1);
  }
  load_frags(f1, xr, wr, DM_ - 32, half);
  mma_frags(acc, f0);
  mma_frags(acc, f1);

  if (mat < 2) {
    bf16* outp = ((mat == 0) ? Qp : Kp) + (size_t)(b * H_ + h) * S_ * DK_;
#pragma unroll
    for (int i = 0; i < 2; i++)
#pragma unroll
      for (int j = 0; j < 4; j++)
#pragma unroll
        for (int r = 0; r < 8; r++) {
          int row = s0 + i * 16 + r + 8 * half, col = j * 16 + ncol;
          outp[(size_t)row * DK_ + col] = f2bf(acc[i][j][r]);
        }
  } else {
    bf16* outp = Vt + (size_t)(b * H_ + h) * DV_ * S_;  // Vt[b,h,v,t]
#pragma unroll
    for (int i = 0; i < 2; i++)
#pragma unroll
      for (int j = 0; j < 4; j++)
#pragma unroll
        for (int r = 0; r < 8; r++) {
          int row = s0 + i * 16 + r + 8 * half, col = j * 16 + ncol;
          outp[(size_t)col * S_ + row] = f2bf(acc[i][j][r]);
        }
  }
}

// ---------------------------------------------------------------- causal flash attention
// One wave per (b,h,16-row query tile). 32-key chunks; K/V staged into
// double-buffered LDS via async-to-LDS, next chunk in flight during compute.
__global__ void __launch_bounds__(32, 1)
k_attn(const bf16* __restrict__ Qp, const bf16* __restrict__ Kp,
       const bf16* __restrict__ Vt, bf16* __restrict__ Ocat) {
  __shared__ bf16 Ksh[2][32 * 64]; // K chunk: 32 rows(t) x 64(dk)
  __shared__ bf16 Vsh[2][64 * 32]; // V^T chunk: 64 rows(v) x 32(t)
  __shared__ bf16 Pl[16 * 32];     // P tile, C-layout -> A-layout staging

  int bid = blockIdx.x;
  int stile = bid & 63; bid >>= 6;
  int h = bid & 15;
  int b = bid >> 4;

  int lane = threadIdx.x, half = lane >> 4, ncol = lane & 15;
  int s0 = stile * 16;

  const bf16* Qb = Qp + (size_t)(b * H_ + h) * S_ * DK_;
  const bf16* Kb = Kp + (size_t)(b * H_ + h) * S_ * DK_;
  const bf16* Vb = Vt + (size_t)(b * H_ + h) * DV_ * S_;

  auto stage_K = [&](bf16* kbuf, const bf16* kg) {   // 4KB contiguous
    unsigned base = lds_off_of(kbuf);
#pragma unroll
    for (int i = 0; i < 8; i++)
      async_ld16(base + i * 512 + lane * 16, kg + i * 256 + lane * 8);
  };
  auto stage_V = [&](bf16* vbuf, const bf16* vg) {   // 64 rows x 64B, S-strided
    unsigned base = lds_off_of(vbuf);
#pragma unroll
    for (int i = 0; i < 8; i++) {
      int p = i * 32 + lane;
      int vrow = p >> 2, q = p & 3;
      async_ld16(base + vrow * 64 + q * 16, vg + (size_t)vrow * S_ + q * 8);
    }
  };

  const bf16* qrow = Qb + (size_t)(s0 + ncol) * DK_;
  v16bf aq0 = fragA(qrow, half);       // k = 0..31
  v16bf aq1 = fragA(qrow + 32, half);  // k = 32..63

  float m_i[8], l_i[8];
#pragma unroll
  for (int r = 0; r < 8; r++) { m_i[r] = -3.0e38f; l_i[r] = 0.0f; }
  v8f o0 = {}, o1 = {}, o2 = {}, o3 = {};
  const float scale = 0.125f; // 1/sqrt(64)

  stage_K(Ksh[0], Kb);
  stage_V(Vsh[0], Vb);
  int buf = 0;

  for (int t0 = 0; t0 < s0 + 16; t0 += 32) {
    bool has_next = (t0 + 32 < s0 + 16);
    if (has_next) {
      stage_K(Ksh[buf ^ 1], Kb + (size_t)(t0 + 32) * DK_);
      stage_V(Vsh[buf ^ 1], Vb + (t0 + 32));
      wait_async_16();   // 16 newer ops outstanding -> current chunk complete
    } else {
      wait_async_all();
    }
    const bf16* kl = Ksh[buf];
    const bf16* vl = Vsh[buf];

    const bf16* kr0 = kl + (size_t)ncol * 64;
    const bf16* kr1 = kl + (size_t)(16 + ncol) * 64;
    v8f sc0 = {}, sc1 = {};
    sc0 = wmma_bf16(aq0, fragB(kr0, half), sc0);
    sc0 = wmma_bf16(aq1, fragB(kr0 + 32, half), sc0);
    sc1 = wmma_bf16(aq0, fragB(kr1, half), sc1);
    sc1 = wmma_bf16(aq1, fragB(kr1 + 32, half), sc1);

    float p0[8], p1[8];
#pragma unroll
    for (int r = 0; r < 8; r++) {
      int srow = s0 + r + 8 * half;
      float v0 = sc0[r] * scale; if (t0 + ncol > srow)      v0 = -1.0e30f;
      float v1 = sc1[r] * scale; if (t0 + 16 + ncol > srow) v1 = -1.0e30f;
      p0[r] = v0; p1[r] = v1;
    }
    float mr[8];
#pragma unroll
    for (int r = 0; r < 8; r++) mr[r] = fmaxf(p0[r], p1[r]);
#pragma unroll
    for (int off = 1; off < 16; off <<= 1)
#pragma unroll
      for (int r = 0; r < 8; r++) mr[r] = fmaxf(mr[r], __shfl_xor(mr[r], off, 32));

    float fs[8];
#pragma unroll
    for (int r = 0; r < 8; r++) {
      float mn = fmaxf(m_i[r], mr[r]);
      fs[r] = __expf(m_i[r] - mn);
      m_i[r] = mn;
      p0[r] = __expf(p0[r] - mn);
      p1[r] = __expf(p1[r] - mn);
    }
    float rs[8];
#pragma unroll
    for (int r = 0; r < 8; r++) rs[r] = p0[r] + p1[r];
#pragma unroll
    for (int off = 1; off < 16; off <<= 1)
#pragma unroll
      for (int r = 0; r < 8; r++) rs[r] += __shfl_xor(rs[r], off, 32);

#pragma unroll
    for (int r = 0; r < 8; r++) {
      l_i[r] = l_i[r] * fs[r] + rs[r];
      o0[r] *= fs[r]; o1[r] *= fs[r]; o2[r] *= fs[r]; o3[r] *= fs[r];
      Pl[(r + 8 * half) * 32 + ncol]      = f2bf(p0[r]);
      Pl[(r + 8 * half) * 32 + 16 + ncol] = f2bf(p1[r]);
    }
    __syncthreads(); // single-wave WG: DS in-order; barrier lowers to NOP

    v16bf ap = fragA(&Pl[ncol * 32], half); // P (16s x 32t) as A-operand
    o0 = wmma_bf16(ap, fragB(vl + (size_t)(0 * 16 + ncol) * 32, half), o0);
    o1 = wmma_bf16(ap, fragB(vl + (size_t)(1 * 16 + ncol) * 32, half), o1);
    o2 = wmma_bf16(ap, fragB(vl + (size_t)(2 * 16 + ncol) * 32, half), o2);
    o3 = wmma_bf16(ap, fragB(vl + (size_t)(3 * 16 + ncol) * 32, half), o3);
    __syncthreads();
    buf ^= 1;
  }

  v8f oo[4] = {o0, o1, o2, o3};
#pragma unroll
  for (int r = 0; r < 8; r++) {
    float inv = 1.0f / l_i[r];
    int srow = s0 + r + 8 * half;
    bf16* orow = Ocat + (size_t)(b * S_ + srow) * E_ + h * DV_;
#pragma unroll
    for (int j = 0; j < 4; j++) orow[j * 16 + ncol] = f2bf(oo[j][r] * inv);
  }
}

// ---------------------------------------------------------------- output projection
// y = X @ Wo^T ; Wo row-major [DM,E] is already B^T-row-major. 32x64 tiles.
__global__ void __launch_bounds__(32, 1)
k_outproj(const bf16* __restrict__ X, const bf16* __restrict__ Wob,
          float* __restrict__ Y) {
  int bid = blockIdx.x;
  int rowtile = bid & 127; // B*S/32 = 128
  int dtile = bid >> 7;    // DM/64 = 16

  int lane = threadIdx.x, half = lane >> 4, ncol = lane & 15;
  int g0 = rowtile * 32, d0 = dtile * 64;

  const bf16* xr[2];
  const bf16* wr[4];
#pragma unroll
  for (int i = 0; i < 2; i++) xr[i] = X + (size_t)(g0 + i * 16 + ncol) * E_;
#pragma unroll
  for (int j = 0; j < 4; j++) wr[j] = Wob + (size_t)(d0 + j * 16 + ncol) * E_;

  __builtin_prefetch(xr[0] + 512, 0, 0);
  __builtin_prefetch(wr[0] + 512, 0, 0);

  v8f acc[2][4];
#pragma unroll
  for (int i = 0; i < 2; i++)
#pragma unroll
    for (int j = 0; j < 4; j++) acc[i][j] = {};

  Frags f0, f1;
  load_frags(f0, xr, wr, 0, half);
  for (int e = 0; e < E_ - 64; e += 64) {
    load_frags(f1, xr, wr, e + 32, half);
    mma_frags(acc, f0);
    load_frags(f0, xr, wr, e + 64, half);
    mma_frags(acc, f1);
  }
  load_frags(f1, xr, wr, E_ - 32, half);
  mma_frags(acc, f0);
  mma_frags(acc, f1);

#pragma unroll
  for (int i = 0; i < 2; i++)
#pragma unroll
    for (int j = 0; j < 4; j++)
#pragma unroll
      for (int r = 0; r < 8; r++)
        Y[(size_t)(g0 + i * 16 + r + 8 * half) * DM_ + d0 + j * 16 + ncol] =
            acc[i][j][r];
}

// ---------------------------------------------------------------- launch
extern "C" void kernel_launch(void* const* d_in, const int* in_sizes, int n_in,
                              void* d_out, int out_size, void* d_ws, size_t ws_size,
                              hipStream_t stream) {
  const float* q  = (const float*)d_in[0];
  const float* k  = (const float*)d_in[1];
  const float* v  = (const float*)d_in[2];
  const float* Wq = (const float*)d_in[3];
  const float* Wk = (const float*)d_in[4];
  const float* Wv = (const float*)d_in[5];
  const float* Wo = (const float*)d_in[6];
  float* y = (float*)d_out;

  char* ws = (char*)d_ws;
  size_t off = 0;
  auto carve = [&](size_t elems) -> bf16* {
    bf16* p = (bf16*)(ws + off);
    off += ((elems * sizeof(bf16) + 255) & ~(size_t)255);
    return p;
  };
  bf16* qb   = carve((size_t)B_ * S_ * DM_);
  bf16* kb   = carve((size_t)B_ * S_ * DM_);
  bf16* vb   = carve((size_t)B_ * S_ * DM_);
  bf16* Wqt  = carve((size_t)H_ * DK_ * DM_);
  bf16* Wkt  = carve((size_t)H_ * DK_ * DM_);
  bf16* Wvt  = carve((size_t)H_ * DV_ * DM_);
  bf16* Wob  = carve((size_t)DM_ * E_);
  bf16* Qp   = carve((size_t)B_ * H_ * S_ * DK_);
  bf16* Kp   = carve((size_t)B_ * H_ * S_ * DK_);
  bf16* Vtb  = carve((size_t)B_ * H_ * DV_ * S_);
  bf16* Ocat = carve((size_t)B_ * S_ * E_);

  int nx = B_ * S_ * DM_;
  int nw = H_ * DM_ * DK_;
  k_cvt<<<(nx + 255) / 256, 256, 0, stream>>>(q, qb, nx);
  k_cvt<<<(nx + 255) / 256, 256, 0, stream>>>(k, kb, nx);
  k_cvt<<<(nx + 255) / 256, 256, 0, stream>>>(v, vb, nx);
  k_cvt<<<(DM_ * E_ + 255) / 256, 256, 0, stream>>>(Wo, Wob, DM_ * E_);
  k_transpose_w<<<(nw + 255) / 256, 256, 0, stream>>>(Wq, Wqt);
  k_transpose_w<<<(nw + 255) / 256, 256, 0, stream>>>(Wk, Wkt);
  k_transpose_w<<<(nw + 255) / 256, 256, 0, stream>>>(Wv, Wvt);

  k_qkv_proj<<<3 * B_ * H_ * (S_ / 32), 32, 0, stream>>>(qb, kb, vb, Wqt, Wkt, Wvt,
                                                         Qp, Kp, Vtb);
  k_attn<<<B_ * H_ * (S_ / 16), 32, 0, stream>>>(Qp, Kp, Vtb, Ocat);
  k_outproj<<<(B_ * S_ / 32) * (DM_ / 64), 32, 0, stream>>>(Ocat, Wob, y);
}
